// V2VNetFusion_81054622810364
// MI455X (gfx1250) — compile-verified, matrix operand order checked by reading, not run on hardware
//
#include <hip/hip_runtime.h>
#include <hip/hip_bf16.h>
#include <stdint.h>

// Problem constants
#define HH 200
#define WW 352
#define HWSZ (HH * WW)      // 70400
#define CC 64
#define BB 4

typedef __attribute__((ext_vector_type(16))) __bf16 v16bf;
typedef __attribute__((ext_vector_type(8)))  float  v8f;

// ---------- bf16 helpers (bit-exact RNE, storage as ushort) ----------
__device__ __forceinline__ unsigned short f2bf(float f) {
    union { float f; uint32_t u; } c; c.f = f;
    uint32_t u = c.u;
    uint32_t r = (u + 0x7FFFu + ((u >> 16) & 1u)) >> 16;
    return (unsigned short)r;
}
__device__ __forceinline__ float bf2f(unsigned short b) {
    union { uint32_t u; float f; } c; c.u = ((uint32_t)b) << 16;
    return c.f;
}
__device__ __forceinline__ float ldval(const float* p) { return *p; }
__device__ __forceinline__ float ldval(const unsigned short* p) { return bf2f(*p); }

// ---------- pre-kernels ----------
// Permute (Cout, Cin, 3, 3) f32 -> bf16 with K reordered kernel-position-major:
//   dst[co*K + r*Cin + ci] = src[(co*Cin + ci)*9 + r],  K = Cin*9
__global__ void cvt_w_kernel(const float* __restrict__ src,
                             unsigned short* __restrict__ dst,
                             int Cout, int Cin) {
    int i = blockIdx.x * blockDim.x + threadIdx.x;
    int total = Cout * Cin * 9;
    if (i >= total) return;
    int co  = i / (Cin * 9);
    int rem = i - co * (Cin * 9);
    int ci  = rem / 9;
    int r   = rem - ci * 9;
    dst[(size_t)co * (Cin * 9) + r * Cin + ci] = f2bf(src[i]);
}

// cat_feat[b][0..63] = bf16(x0[b])   (cat_feat layout: (B,128,HW) bf16)
__global__ void seed_cat_kernel(const float* __restrict__ x0,
                                unsigned short* __restrict__ catb) {
    int i = blockIdx.x * blockDim.x + threadIdx.x;   // grid sized exactly B*64*HW
    int b   = i / (CC * HWSZ);
    int rem = i - b * (CC * HWSZ);
    catb[(size_t)(b * 2 * CC) * HWSZ + rem] = f2bf(x0[i]);
}

// ---------- implicit-GEMM 3x3 conv tile core (WMMA bf16, f32 acc) ----------
// Output tile: (NWAVES*16) Cout x 32 pixels (one row strip, y fixed).
// Stage-once shifted-window: the 3 x 34 x CIN input halo tile is staged into
// LDS once (single barrier); each (ky,kx) tap's B fragment is a shifted LDS
// read of that tile. K loop is a barrier-free stream of loads + WMMAs.
// Weights are pre-permuted r-major (k = r*CIN + ci), bf16.
template <typename T, int C0, int CIN, int NWAVES>
__device__ __forceinline__ void conv_tile(
    const T* __restrict__ src0, const T* __restrict__ src1, int src1_zero,
    const unsigned short* __restrict__ wb,
    v8f acc[2], int y, int xbase)
{
    constexpr int CP = CIN + 8;                 // pad: (CP*2) % 16 == 0, banks rotate
    __shared__ unsigned short S[3][34][CP];

    const int tid  = threadIdx.x;
    const int lane = tid & 31;
    const int wave = tid >> 5;
    const int half = lane >> 4;     // K-halving per documented 16-bit layout
    const int lrow = lane & 15;
    constexpr int NT = NWAVES * 32;

    #pragma unroll
    for (int q = 0; q < 8; ++q) { acc[0][q] = 0.0f; acc[1][q] = 0.0f; }

    // ---- stage 3 x 34 x CIN halo tile (zero-padded borders), one barrier ----
    // idx = c*102 + e, e = ry*34 + px: consecutive tids -> consecutive px
    // (coalesced global loads); LDS store stride CP*2 rotates banks.
    constexpr int TOT = 3 * 34 * CIN;
    for (int idx = tid; idx < TOT; idx += NT) {
        int c  = idx / 102;
        int e  = idx - c * 102;
        int ry = e / 34;
        int px = e - ry * 34;
        int iy = y + ry - 1;
        int ix = xbase + px - 1;
        bool inb = ((unsigned)iy < (unsigned)HH) && ((unsigned)ix < (unsigned)WW);
        if (src1_zero && c >= C0) inb = false;
        int moff = inb ? (iy * WW + ix) : 0;         // clamped address, masked value
        const T* p = (c < C0) ? (src0 + (size_t)c * HWSZ + moff)
                              : (src1 + (size_t)(c - C0) * HWSZ + moff);
        float v = ldval(p);
        S[ry][px][c] = f2bf(inb ? v : 0.0f);
    }
    __syncthreads();

    // ---- K loop: 9 taps x (CIN/32) chunks, no barriers ----
    constexpr int K = CIN * 9;
    const unsigned short* wrow = wb + (size_t)(wave * 16 + lrow) * K;

    int k0 = 0;
    for (int r = 0; r < 9; ++r) {
        const int ky = (r >= 6) ? 2 : (r >= 3) ? 1 : 0;
        const int kx = r - ky * 3;
        // per-lane fragment base rows (16B-aligned: CP*2 and half*16 both x16)
        const unsigned short* row0 = &S[ky][lrow + kx][half * 8];
        const unsigned short* row1 = &S[ky][16 + lrow + kx][half * 8];
        #pragma unroll
        for (int cb = 0; cb < CIN / 32; ++cb, k0 += 32) {
            union Frag { v16bf v; uint4 u[2]; } fa, fb0, fb1;
            fa.u[0]  = *(const uint4*)(wrow + k0 + half * 8);
            fa.u[1]  = *(const uint4*)(wrow + k0 + 16 + half * 8);
            fb0.u[0] = *(const uint4*)(row0 + cb * 32);
            fb0.u[1] = *(const uint4*)(row0 + cb * 32 + 16);
            fb1.u[0] = *(const uint4*)(row1 + cb * 32);
            fb1.u[1] = *(const uint4*)(row1 + cb * 32 + 16);

            acc[0] = __builtin_amdgcn_wmma_f32_16x16x32_bf16(
                false, fa.v, false, fb0.v, (short)0, acc[0], false, false);
            acc[1] = __builtin_amdgcn_wmma_f32_16x16x32_bf16(
                false, fa.v, false, fb1.v, (short)0, acc[1], false, false);
        }
    }
}

// C/D layout: lane(0..15)=col n, lane(16..31)=col n (+8 rows); VGPR q -> row q (+8 hi half)
#define EPILOGUE_COORDS \
    const int lane = threadIdx.x & 31; \
    const int wave = threadIdx.x >> 5; \
    const int nl   = lane & 15;        \
    const int mh   = lane >> 4;

// ---------- conv 1: message + agg, fused into cat_feat[64..127] ----------
__global__ void __launch_bounds__(128) conv_msg_kernel(
    const float* __restrict__ x,                 // (2,B,64,H,W)
    const unsigned short* __restrict__ wbm,      // (64,1152) bf16, r-major K
    const float* __restrict__ msg_b,
    unsigned short* __restrict__ catb)           // (B,128,HW) bf16
{
    const int b     = blockIdx.z;
    const int y     = blockIdx.y;
    const int xbase = blockIdx.x * 32;
    const float* s0 = x + (size_t)((0 * BB + b) * CC) * HWSZ;
    const float* s1 = x + (size_t)((1 * BB + b) * CC) * HWSZ;

    v8f acc[2];
    conv_tile<float, CC, 2 * CC, 4>(s0, s1, 0, wbm, acc, y, xbase);

    EPILOGUE_COORDS
    #pragma unroll
    for (int ns = 0; ns < 2; ++ns)
        #pragma unroll
        for (int q = 0; q < 8; ++q) {
            int co  = wave * 16 + mh * 8 + q;
            int pos = y * WW + xbase + ns * 16 + nl;
            float msg = acc[ns][q] + msg_b[co];
            float x0v = s0[(size_t)co * HWSZ + pos];
            catb[(size_t)(b * 2 * CC + CC + co) * HWSZ + pos] = f2bf(0.5f * (x0v + msg));
        }
}

// ---------- conv 2: GRU gates -> reset*h (bf16) and update (f32) ----------
__global__ void __launch_bounds__(256) conv_gates_kernel(
    const unsigned short* __restrict__ catb_b,   // (128,HW) bf16
    const unsigned short* __restrict__ hb,       // (64,HW) bf16 (prev h)
    int h_zero,
    const unsigned short* __restrict__ wbg,      // (128,1728) bf16, r-major K
    const float* __restrict__ gates_b,
    const float* __restrict__ h_prev,            // (64,HW) f32 (prev out slice)
    unsigned short* __restrict__ rhb,            // (64,HW) bf16
    float* __restrict__ upd)                     // (64,HW) f32
{
    const int y     = blockIdx.y;
    const int xbase = blockIdx.x * 32;

    v8f acc[2];
    conv_tile<unsigned short, 2 * CC, 3 * CC, 8>(
        catb_b, hb, h_zero, wbg, acc, y, xbase);

    EPILOGUE_COORDS
    #pragma unroll
    for (int ns = 0; ns < 2; ++ns)
        #pragma unroll
        for (int q = 0; q < 8; ++q) {
            int co  = wave * 16 + mh * 8 + q;
            int pos = y * WW + xbase + ns * 16 + nl;
            float g = 1.0f / (1.0f + __expf(-(acc[ns][q] + gates_b[co])));
            if (co < CC) {
                float h = h_zero ? 0.0f : h_prev[(size_t)co * HWSZ + pos];
                rhb[(size_t)co * HWSZ + pos] = f2bf(g * h);
            } else {
                upd[(size_t)(co - CC) * HWSZ + pos] = g;
            }
        }
}

// ---------- conv 3: candidate + GRU blend -> out[b] (f32) and hb (bf16) ----------
__global__ void __launch_bounds__(128) conv_can_kernel(
    const unsigned short* __restrict__ catb_b,   // (128,HW) bf16
    const unsigned short* __restrict__ rhb,      // (64,HW) bf16 (reset*h)
    int h_zero,
    const unsigned short* __restrict__ wbc,      // (64,1728) bf16, r-major K
    const float* __restrict__ can_b,
    const float* __restrict__ upd,               // (64,HW) f32
    const float* __restrict__ h_prev,            // (64,HW) f32
    float* __restrict__ out_b,                   // (64,HW) f32
    unsigned short* __restrict__ hb)             // (64,HW) bf16 (next h, conv input)
{
    const int y     = blockIdx.y;
    const int xbase = blockIdx.x * 32;

    v8f acc[2];
    conv_tile<unsigned short, 2 * CC, 3 * CC, 4>(
        catb_b, rhb, 0, wbc, acc, y, xbase);

    EPILOGUE_COORDS
    #pragma unroll
    for (int ns = 0; ns < 2; ++ns)
        #pragma unroll
        for (int q = 0; q < 8; ++q) {
            int co  = wave * 16 + mh * 8 + q;
            int pos = y * WW + xbase + ns * 16 + nl;
            float cn = tanhf(acc[ns][q] + can_b[co]);
            float u  = upd[(size_t)co * HWSZ + pos];
            float h  = h_zero ? 0.0f : h_prev[(size_t)co * HWSZ + pos];
            float hn = (1.0f - u) * h + u * cn;
            out_b[(size_t)co * HWSZ + pos] = hn;
            hb[(size_t)co * HWSZ + pos]    = f2bf(hn);
        }
}

// ---------- launcher ----------
extern "C" void kernel_launch(void* const* d_in, const int* in_sizes, int n_in,
                              void* d_out, int out_size, void* d_ws, size_t ws_size,
                              hipStream_t stream) {
    const float* x       = (const float*)d_in[0];  // (2,4,64,200,352)
    const float* msg_w   = (const float*)d_in[1];  // (64,128,3,3)
    const float* msg_b   = (const float*)d_in[2];
    const float* gates_w = (const float*)d_in[3];  // (128,192,3,3)
    const float* gates_b = (const float*)d_in[4];
    const float* can_w   = (const float*)d_in[5];  // (64,192,3,3)
    const float* can_b   = (const float*)d_in[6];
    float* out = (float*)d_out;                    // (4,64,200,352)

    // workspace carve (256B aligned)
    char* ws = (char*)d_ws;
    auto carve = [&](size_t bytes) {
        char* p = ws;
        ws += (bytes + 255) & ~(size_t)255;
        return p;
    };
    const int NW_MSG = 64 * 128 * 9;    // 73728
    const int NW_G   = 128 * 192 * 9;   // 221184
    const int NW_C   = 64 * 192 * 9;    // 110592
    unsigned short* wbm  = (unsigned short*)carve((size_t)NW_MSG * 2);
    unsigned short* wbg  = (unsigned short*)carve((size_t)NW_G * 2);
    unsigned short* wbc  = (unsigned short*)carve((size_t)NW_C * 2);
    unsigned short* catb = (unsigned short*)carve((size_t)BB * 2 * CC * HWSZ * 2);
    unsigned short* hb   = (unsigned short*)carve((size_t)CC * HWSZ * 2);
    unsigned short* rhb  = (unsigned short*)carve((size_t)CC * HWSZ * 2);
    float*          upd  = (float*)carve((size_t)CC * HWSZ * 4);

    // weights -> bf16, K permuted r-major
    cvt_w_kernel<<<(NW_MSG + 255) / 256, 256, 0, stream>>>(msg_w, wbm, 64, 128);
    cvt_w_kernel<<<(NW_G   + 255) / 256, 256, 0, stream>>>(gates_w, wbg, 128, 192);
    cvt_w_kernel<<<(NW_C   + 255) / 256, 256, 0, stream>>>(can_w, wbc, 64, 192);

    // cat_feat[:, 0:64] = bf16(x0)   (x0 is the leading B*64*HW block of x)
    seed_cat_kernel<<<(BB * CC * HWSZ) / 256, 256, 0, stream>>>(x, catb);

    // message conv + agg, fused into cat_feat[:, 64:128]
    {
        dim3 grid(WW / 32, HH, BB);
        conv_msg_kernel<<<grid, 128, 0, stream>>>(x, wbm, msg_b, catb);
    }

    // sequential ConvGRU over B=4 steps; h_prev read from out[b-1]
    for (int b = 0; b < BB; ++b) {
        const unsigned short* catb_b = catb + (size_t)b * 2 * CC * HWSZ;
        const float* h_prev = (b == 0) ? out : out + (size_t)(b - 1) * CC * HWSZ;
        int h_zero = (b == 0) ? 1 : 0;
        dim3 grid(WW / 32, HH, 1);
        conv_gates_kernel<<<grid, 256, 0, stream>>>(
            catb_b, hb, h_zero, wbg, gates_b, h_prev, rhb, upd);
        conv_can_kernel<<<grid, 128, 0, stream>>>(
            catb_b, rhb, h_zero, wbc, can_b, upd, h_prev,
            out + (size_t)b * CC * HWSZ, hb);
    }
}
